// Topk_noisy_Router_8512625180882
// MI455X (gfx1250) — matrix operand order repaired and checked by reading, not compiled, not run on hardware
//
#include <hip/hip_runtime.h>
#include <hip/hip_bf16.h>
#include <math.h>

typedef __attribute__((ext_vector_type(16))) __bf16 v16bf;
typedef __attribute__((ext_vector_type(8)))  float  v8f;
typedef __attribute__((ext_vector_type(4)))  unsigned int u32x4;

// Problem constants (fixed by the reference)
constexpr int D    = 2048;   // hidden dim (K)
constexpr int E    = 64;     // experts
constexpr int NB   = 128;    // combined output cols: [router(64) ; noise(64)]
constexpr int BM   = 128;    // tokens per block
constexpr int BK   = 32;     // K per WMMA step
constexpr int KIT  = D / BK; // 64 K-iterations
constexpr int XPAD = 40;     // bf16 per LDS row (32 data + 8 pad) -> 80B rows, conflict-free b128
constexpr int APAD = 132;    // f32 per accum row (128 data + 4 pad)

// LDS layout (bytes)
constexpr int SX_OFF   = 0;                       // 2 * BM * XPAD * 2  = 20480
constexpr int SW_OFF   = 2 * BM * XPAD * 2;       // 20480
constexpr int SACC_OFF = SW_OFF + 2 * NB * XPAD * 2;              // 40960
constexpr int SB_OFF   = SACC_OFF + BM * APAD * 4;                // 108544
constexpr int SMEM_BYTES = SB_OFF + NB * 4;                       // 109056

__global__ __launch_bounds__(256)
void moe_router_topk_kernel(const float* __restrict__ x,
                            const float* __restrict__ rw,
                            const float* __restrict__ rb,
                            const float* __restrict__ nw,
                            const float* __restrict__ nb,
                            const float* __restrict__ nu,
                            float* __restrict__ outP,
                            float* __restrict__ outI,
                            int tokens)
{
    extern __shared__ __align__(16) char smem[];
    __bf16* sX   = (__bf16*)(smem + SX_OFF);    // [2][BM][XPAD]
    __bf16* sW   = (__bf16*)(smem + SW_OFF);    // [2][NB][XPAD]
    float*  sAcc = (float*) (smem + SACC_OFF);  // [BM][APAD]
    float*  sBias= (float*) (smem + SB_OFF);    // [NB]

    const int tid  = threadIdx.x;
    const int wave = tid >> 5;
    const int lane = tid & 31;
    const int nIdx = lane & 15;   // M (A) / N (B,C) within tile
    const int hi   = lane >> 4;   // K-half selector

    // wave tile block: 4 row-tiles x 2 col-tiles over the 8x8 tile grid
    const int waveR = wave >> 2;  // 0..1 -> row tiles [waveR*4, +4)
    const int waveC = wave & 3;   // 0..3 -> col tiles [waveC*2, +2)

    const int rowBlock = blockIdx.x * BM;

    // stage the combined bias vector once
    if (tid < NB) sBias[tid] = (tid < E) ? rb[tid] : nb[tid - E];

    // ---- global->reg->LDS staging assignment: thread covers one row-half ----
    const int gRow  = tid >> 1;   // 0..127 : token row AND combined-expert row
    const int gHalf = tid & 1;    // which 16-float half of the 32-float K chunk
    const float* xBase = x + (size_t)(rowBlock + gRow) * D + gHalf * 16;
    const float* wBase = (gRow < E ? rw + (size_t)gRow * D
                                   : nw + (size_t)(gRow - E) * D) + gHalf * 16;

    float fx[16], fw[16];
    auto loadChunk = [&](int k) {
        const float4* px = (const float4*)(xBase + k * BK);
        const float4* pw = (const float4*)(wBase + k * BK);
        #pragma unroll
        for (int j = 0; j < 4; ++j) {
            float4 a = px[j];
            fx[4*j+0] = a.x; fx[4*j+1] = a.y; fx[4*j+2] = a.z; fx[4*j+3] = a.w;
            float4 b = pw[j];
            fw[4*j+0] = b.x; fw[4*j+1] = b.y; fw[4*j+2] = b.z; fw[4*j+3] = b.w;
        }
    };
    auto storeChunk = [&](int buf) {
        union { __bf16 h[16]; u32x4 q[2]; } cx, cw;
        #pragma unroll
        for (int j = 0; j < 16; ++j) {
            cx.h[j] = (__bf16)fx[j];
            cw.h[j] = (__bf16)fw[j];
        }
        u32x4* dx = (u32x4*)&sX[(size_t)buf * BM * XPAD + gRow * XPAD + gHalf * 16];
        dx[0] = cx.q[0]; dx[1] = cx.q[1];
        u32x4* dw = (u32x4*)&sW[(size_t)buf * NB * XPAD + gRow * XPAD + gHalf * 16];
        dw[0] = cw.q[0]; dw[1] = cw.q[1];
    };

    // ---- accumulators: acc[c*4 + r] for (row tile waveR*4+r, col tile waveC*2+c) ----
    v8f acc[8];
    #pragma unroll
    for (int i = 0; i < 8; ++i)
        acc[i] = v8f{0.f,0.f,0.f,0.f,0.f,0.f,0.f,0.f};

    union Frag { v16bf v; u32x4 q[2]; };

    auto compute = [&](int buf) {
        // Load all 4 A fragments first: they feed BOTH c-groups of WMMAs, so
        // they must stay live across all 8 WMMAs and cannot be sunk per-use.
        // Each c-group then needs only one 2-load B fragment before its 4
        // back-to-back WMMAs -> 2 dscnt waits per chunk instead of 8.
        Frag A[4];
        #pragma unroll
        for (int r = 0; r < 4; ++r) {
            const __bf16* aP = &sX[(size_t)buf * BM * XPAD
                                   + ((waveR * 4 + r) * 16 + nIdx) * XPAD];
            A[r].q[0] = *(const u32x4*)(aP + hi * 8);       // K 0..7 | 8..15
            A[r].q[1] = *(const u32x4*)(aP + 16 + hi * 8);  // K 16..23 | 24..31
        }
        #pragma unroll
        for (int c = 0; c < 2; ++c) {
            // B 32x16: lane col = nIdx; lanes 0-15 K=0..7,16..23, lanes 16-31 K=8..15,24..31
            const __bf16* bP = &sW[(size_t)buf * NB * XPAD
                                   + ((waveC * 2 + c) * 16 + nIdx) * XPAD + hi * 8];
            Frag B;
            B.q[0] = *(const u32x4*)(bP + 0);
            B.q[1] = *(const u32x4*)(bP + 16);
            #pragma unroll
            for (int r = 0; r < 4; ++r) {
                acc[c * 4 + r] = __builtin_amdgcn_wmma_f32_16x16x32_bf16(
                    false, A[r].v, false, B.v, (short)0, acc[c * 4 + r], false, false);
            }
        }
    };

    // ---- software pipeline: double-buffered LDS, one barrier/iteration ----
    loadChunk(0);
    storeChunk(0);
    __syncthreads();
    for (int k = 0; k < KIT; ++k) {
        const int buf = k & 1;
        if (k + 1 < KIT) loadChunk(k + 1);   // issue next global loads early
        compute(buf);                         // overlap latency with WMMA
        if (k + 1 < KIT) storeChunk(buf ^ 1); // buf^1 reads done at last barrier
        __syncthreads();
    }

    // ---- spill accumulators to LDS: C/D layout VGPR i -> row i + hi*8, col nIdx ----
    #pragma unroll
    for (int c = 0; c < 2; ++c) {
        #pragma unroll
        for (int r = 0; r < 4; ++r) {
            #pragma unroll
            for (int i = 0; i < 8; ++i)
                sAcc[((waveR * 4 + r) * 16 + hi * 8 + i) * APAD
                     + (waveC * 2 + c) * 16 + nIdx] = acc[c * 4 + r][i];
        }
    }
    __syncthreads();

    // ---- epilogue: one thread per token ----
    if (tid < BM) {
        const int tok = rowBlock + tid;
        if (tok < tokens) {
            const float* nuRow = nu + (size_t)tok * E;
            const float* aR = &sAcc[tid * APAD];
            float v1 = -3.0e38f, v2 = -3.0e38f;
            int   i1 = 0, i2 = 0;
            for (int e = 0; e < E; ++e) {
                float lr = aR[e]     + sBias[e];
                float ln = aR[E + e] + sBias[E + e];
                // stable softplus: max(z,0) + log1p(exp(-|z|))
                float sp = fmaxf(ln, 0.f) + log1pf(__expf(-fabsf(ln)));
                float nz = lr + sp * nuRow[e];
                if (nz > v1)      { v2 = v1; i2 = i1; v1 = nz; i1 = e; }
                else if (nz > v2) { v2 = nz; i2 = e; }
            }
            // sparse softmax: exp(NEG_INF - v1) underflows to exactly 0
            float e2  = __expf(v2 - v1);
            float inv = 1.0f / (1.0f + e2);
            float p1 = inv, p2 = e2 * inv;
            float* orow = outP + (size_t)tok * E;
            #pragma unroll
            for (int eb = 0; eb < E; eb += 4) {
                float4 o;
                o.x = (eb + 0 == i1) ? p1 : ((eb + 0 == i2) ? p2 : 0.f);
                o.y = (eb + 1 == i1) ? p1 : ((eb + 1 == i2) ? p2 : 0.f);
                o.z = (eb + 2 == i1) ? p1 : ((eb + 2 == i2) ? p2 : 0.f);
                o.w = (eb + 3 == i1) ? p1 : ((eb + 3 == i2) ? p2 : 0.f);
                *(float4*)(orow + eb) = o;
            }
            outI[(size_t)tok * 2 + 0] = (float)i1;
            outI[(size_t)tok * 2 + 1] = (float)i2;
        }
    }
}

extern "C" void kernel_launch(void* const* d_in, const int* in_sizes, int n_in,
                              void* d_out, int out_size, void* d_ws, size_t ws_size,
                              hipStream_t stream) {
    (void)n_in; (void)out_size; (void)d_ws; (void)ws_size;
    const float* x  = (const float*)d_in[0];
    const float* rw = (const float*)d_in[1];
    const float* rb = (const float*)d_in[2];
    const float* nw = (const float*)d_in[3];
    const float* nb = (const float*)d_in[4];
    const float* nu = (const float*)d_in[5];

    const int tokens = in_sizes[0] / D;        // B*S = 16384
    float* outP = (float*)d_out;               // [tokens, E] probabilities
    float* outI = outP + (size_t)tokens * E;   // [tokens, 2] indices (as float)

    // allow >64KB dynamic LDS (idempotent, deterministic; not a stream op)
    hipFuncSetAttribute((const void*)moe_router_topk_kernel,
                        hipFuncAttributeMaxDynamicSharedMemorySize, SMEM_BYTES);

    const int grid = (tokens + BM - 1) / BM;   // 128 blocks
    hipLaunchKernelGGL(moe_router_topk_kernel, dim3(grid), dim3(256), SMEM_BYTES,
                       stream, x, rw, rb, nw, nb, nu, outP, outI, tokens);
}
